// Quattention_53730040873719
// MI455X (gfx1250) — matrix operand order repaired and checked by reading, not compiled, not run on hardware
//
#include <hip/hip_runtime.h>
#include <hip/hip_bf16.h>

#define BB 4
#define TT 32
#define DD 256
#define NN 8
#define HH 32

typedef __attribute__((ext_vector_type(16))) _Float16 v16h;
typedef __attribute__((ext_vector_type(8)))  float    v8f;

// A-matrix (16x32 f16) lane->K map: lanes 0-15 hold K=0..7,16..23; lanes 16-31 hold K=8..15,24..31
__device__ __forceinline__ int kmapA(int lane, int p) {
  int base = (lane & 16) ? 8 : 0;
  return (p < 8) ? (base + p) : (base + 16 + (p - 8));
}
// B-matrix (32x16 f16) lane->K map: lanes 0-15 hold K=0..7,16..23; lanes 16-31 hold K=8..15,24..31
__device__ __forceinline__ int kmapB(int lane, int p) {
  int base = (lane & 16) ? 8 : 0;
  return (p < 8) ? (base + p) : (base + 16 + (p - 8));
}

// LDS byte offset of a __shared__ object (low 32 bits of its generic address)
__device__ __forceinline__ unsigned int ldsOffsetOf(const void* p) {
  return (unsigned int)(uintptr_t)p;
}

// CDNA5 async copy: 16 bytes global -> LDS, tracked by ASYNCcnt (GVS mode).
__device__ __forceinline__ void asyncCopy16(unsigned int ldsAddr,
                                            const void* gbase,
                                            unsigned int goff) {
  asm volatile("global_load_async_to_lds_b128 %0, %1, %2"
               :
               : "v"(ldsAddr), "v"(goff), "s"((unsigned long long)(uintptr_t)gbase)
               : "memory");
}
__device__ __forceinline__ void waitAsync0() {
  asm volatile("s_wait_asynccnt 0x0" ::: "memory");
}

// ---------------------------------------------------------------------------
// Kernel 1: projections. X (B*T=128 x D=256) times 7 weight blocks (D x N*H).
// One wave per 16x16 output tile, K-loop of 8 x v_wmma_f32_16x16x32_f16.
// proj layout: [w][b][n][t][h], w in {A,B,C,D,U1,U2,U3}
// ---------------------------------------------------------------------------
__global__ __launch_bounds__(256) void qa_proj_kernel(
    const float* __restrict__ x,
    const float* __restrict__ W_A, const float* __restrict__ W_B,
    const float* __restrict__ W_C, const float* __restrict__ W_D,
    const float* __restrict__ W_DE,
    const float* __restrict__ b_A, const float* __restrict__ b_B,
    const float* __restrict__ b_C, const float* __restrict__ b_D,
    float* __restrict__ proj)
{
  const int lane  = threadIdx.x & 31;
  const int wave  = threadIdx.x >> 5;
  const int tile  = blockIdx.x * 8 + wave;   // 0..895
  const int tileM = tile & 7;                // 8 row tiles over B*T=128
  const int colId = tile >> 3;               // 0..111 = 7 weights * 8 heads * 2 h-tiles
  const int w     = colId >> 4;              // 0..6
  const int rem   = colId & 15;
  const int n     = rem >> 1;
  const int hT    = rem & 1;
  const int m16   = lane & 15;
  const int hcol  = hT * 16 + m16;

  const float* Wbase;
  if (w < 4) {
    const float* Ws = (w == 0) ? W_A : (w == 1) ? W_B : (w == 2) ? W_C : W_D;
    Wbase = Ws + (size_t)n * DD * HH;
  } else {
    Wbase = W_DE + ((size_t)n * 3 + (w - 4)) * DD * HH;  // W_DE: (N, 3D, H)
  }
  // Hint L2: this wave walks the whole (D x H) weight panel.
  __builtin_prefetch(Wbase + (size_t)lane * (DD * HH / 32), 0, 1);

  const int arow = tileM * 16 + m16;         // flattened (b*T + t)
  const float* xrow = x + (size_t)arow * DD;

  v8f acc = {};
  for (int kk = 0; kk < 8; ++kk) {
    v16h a16, b16;
    #pragma unroll
    for (int p = 0; p < 16; ++p) {
      a16[p] = (_Float16)xrow[kk * 32 + kmapA(lane, p)];
      b16[p] = (_Float16)Wbase[(size_t)(kk * 32 + kmapB(lane, p)) * HH + hcol];
    }
    acc = __builtin_amdgcn_wmma_f32_16x16x32_f16(false, a16, false, b16,
                                                 (short)0, acc, false, false);
  }

  float bias = 0.0f;
  if (w < 4) {
    const float* bs = (w == 0) ? b_A : (w == 1) ? b_B : (w == 2) ? b_C : b_D;
    bias = bs[n * HH + hcol];
  }

  const int rowOff = (lane >> 4) << 3;  // D tile: M = r + (lane<16 ? 0 : 8)
  #pragma unroll
  for (int r = 0; r < 8; ++r) {
    int grow = tileM * 16 + rowOff + r;
    int bb2 = grow >> 5, tt2 = grow & 31;
    proj[(((size_t)w * BB + bb2) * NN + n) * (TT * HH) + tt2 * HH + hcol] = acc[r] + bias;
  }
}

// ---------------------------------------------------------------------------
// Kernel 2: fused quartic attention. One wave per (b, n, y).
// scores(t,q)|y,s = [b(t,h)*a(s,h)] x [c(q,h)*d(y,h)]^T  -> 4 WMMAs per s.
// Softmax marginals P1[s], P2[t], P3[q] + denom give z directly:
//   z[h] = (Sum P1*u1 + P2*u2 + P3*u3 + tot*b_DE) / denom
// Operand block staged into LDS with GLOBAL_LOAD_ASYNC_TO_LDS_B128 (ASYNCcnt).
// ---------------------------------------------------------------------------
__global__ __launch_bounds__(256) void qa_attn_kernel(
    const float* __restrict__ proj, const float* __restrict__ b_DE,
    float* __restrict__ zbuf)
{
  const int bn   = blockIdx.x;
  const int bb   = bn / NN, nn = bn % NN;
  const int lane = threadIdx.x & 31;
  const int wave = threadIdx.x >> 5;          // 0..7
  const int y    = blockIdx.y * 8 + wave;     // 0..31
  const int m16  = lane & 15;

  __shared__ float sA[TT][HH], sB2[TT][HH], sC[TT][HH], sD[TT][HH];
  __shared__ float sU1[TT][HH], sU2[TT][HH], sU3[TT][HH];
  __shared__ float sP1[8][TT], sP2[8][TT], sP3[8][TT];

  // Async-stage the 7 x 4KB operand block: 256 threads x 16B each per array.
  {
    const size_t wsz  = (size_t)BB * NN * TT * HH;
    const size_t base = ((size_t)bb * NN + nn) * (TT * HH);
    const unsigned int goff = threadIdx.x * 16u;   // byte offset within a 4KB array
    asyncCopy16(ldsOffsetOf(&sA [0][0]) + goff, proj + 0 * wsz + base, goff);
    asyncCopy16(ldsOffsetOf(&sB2[0][0]) + goff, proj + 1 * wsz + base, goff);
    asyncCopy16(ldsOffsetOf(&sC [0][0]) + goff, proj + 2 * wsz + base, goff);
    asyncCopy16(ldsOffsetOf(&sD [0][0]) + goff, proj + 3 * wsz + base, goff);
    asyncCopy16(ldsOffsetOf(&sU1[0][0]) + goff, proj + 4 * wsz + base, goff);
    asyncCopy16(ldsOffsetOf(&sU2[0][0]) + goff, proj + 5 * wsz + base, goff);
    asyncCopy16(ldsOffsetOf(&sU3[0][0]) + goff, proj + 6 * wsz + base, goff);
    waitAsync0();
  }
  __syncthreads();

  const float invScale      = 0.005524271728019903f;   // 1 / 32^1.5
  const float SCALED_IGNORE = -1.0e6f * invScale;

  // Stage per-lane operand pieces. d(y,h) folded into the B side only.
  float bv0[16], bv1[16];
  v16h cop0, cop1;
  #pragma unroll
  for (int p = 0; p < 16; ++p) {
    int ka = kmapA(lane, p);
    bv0[p] = sB2[m16][ka];
    bv1[p] = sB2[m16 + 16][ka];
    int kb = kmapB(lane, p);
    float dk = sD[y][kb];
    cop0[p] = (_Float16)(sC[m16][kb] * dk);
    cop1[p] = (_Float16)(sC[m16 + 16][kb] * dk);
  }

  for (int i = lane; i < TT; i += 32) {
    sP1[wave][i] = 0.0f; sP2[wave][i] = 0.0f; sP3[wave][i] = 0.0f;
  }

  const int rowOff = (lane >> 4) << 3;
  const int tMax = (y > 16) ? 2 : 1;   // tiles with t>=y (or q>=y) are fully masked
  const int qMax = (y > 16) ? 2 : 1;   // (wave-uniform -> EXEC stays all-ones)

  float m = -1000.0f;                  // PAD logit is always in the softmax
  float sumE = 0.0f;
  float p3part0 = 0.0f, p3part1 = 0.0f;
  float p2part[2][8] = {};

  for (int pass = 0; pass < 2; ++pass) {
    if (pass == 1) {
      #pragma unroll
      for (int off = 16; off > 0; off >>= 1)
        m = fmaxf(m, __shfl_xor(m, off, 32));
    }
    for (int s = 0; s < y; ++s) {      // s >= y fully masked -> skipped
      v16h aop0, aop1;                 // A[t,h] = b[t,h]*a[s,h]
      #pragma unroll
      for (int p = 0; p < 16; ++p) {
        float av = sA[s][kmapA(lane, p)];
        aop0[p] = (_Float16)(bv0[p] * av);
        aop1[p] = (_Float16)(bv1[p] * av);
      }
      float p1loc = 0.0f;
      for (int tM = 0; tM < tMax; ++tM) {
        v16h aop = aop0; if (tM) aop = aop1;
        for (int qN = 0; qN < qMax; ++qN) {
          v16h cop = cop0; if (qN) cop = cop1;
          v8f accD = {};
          accD = __builtin_amdgcn_wmma_f32_16x16x32_f16(false, aop, false, cop,
                                                        (short)0, accD, false, false);
          const int q = qN * 16 + m16;
          #pragma unroll
          for (int r = 0; r < 8; ++r) {
            const int t = tM * 16 + rowOff + r;
            const bool masked = (t >= y) || (q >= y) || (s == t) || (s == q) || (t == q);
            const float val = masked ? SCALED_IGNORE : accD[r] * invScale;
            if (pass == 0) {
              if (!masked) m = fmaxf(m, val);
            } else {
              const float e = masked ? 0.0f : __expf(val - m);
              sumE  += e;
              p1loc += e;
              if (qN) p3part1 += e; else p3part0 += e;
              p2part[tM][r] += e;
            }
          }
        }
      }
      if (pass == 1) {
        #pragma unroll
        for (int off = 16; off > 0; off >>= 1)
          p1loc += __shfl_xor(p1loc, off, 32);
        if (lane == 0) sP1[wave][s] = p1loc;
      }
    }
  }

  float tot = sumE;
  #pragma unroll
  for (int off = 16; off > 0; off >>= 1) tot += __shfl_xor(tot, off, 32);
  const float denom = tot + __expf(-1000.0f - m);   // + pad column

  // P3[q]: combine the two M-halves (lanes l and l^16 share a q column)
  p3part0 += __shfl_xor(p3part0, 16, 32);
  p3part1 += __shfl_xor(p3part1, 16, 32);
  sP3[wave][lane] = (lane < 16) ? p3part0 : p3part1;

  // P2[t]: reduce over the 16 q-lanes of each half
  #pragma unroll
  for (int tM = 0; tM < 2; ++tM) {
    #pragma unroll
    for (int r = 0; r < 8; ++r) {
      float v = p2part[tM][r];
      v += __shfl_xor(v, 1, 32);
      v += __shfl_xor(v, 2, 32);
      v += __shfl_xor(v, 4, 32);
      v += __shfl_xor(v, 8, 32);
      if (m16 == 0) sP2[wave][tM * 16 + rowOff + r] = v;
    }
  }

  // z via marginals (lane == h)
  const int h = lane;
  float accz = tot * b_DE[nn * HH + h];
  for (int i = 0; i < TT; ++i) {
    accz += sP1[wave][i] * sU1[i][h]
          + sP2[wave][i] * sU2[i][h]
          + sP3[wave][i] * sU3[i][h];
  }
  zbuf[(((size_t)bb * TT + y) * NN + nn) * HH + h] = accz / denom;
}

// ---------------------------------------------------------------------------
// Kernel 3: out = z (128 x 256) * W_O (N*H=256 x D=256) + b_O, WMMA tiles.
// ---------------------------------------------------------------------------
__global__ __launch_bounds__(256) void qa_out_kernel(
    const float* __restrict__ zbuf, const float* __restrict__ W_O,
    const float* __restrict__ b_O, float* __restrict__ out)
{
  const int lane  = threadIdx.x & 31;
  const int wave  = threadIdx.x >> 5;
  const int tile  = blockIdx.x * 8 + wave;  // 0..127
  const int tileM = tile & 7;
  const int tileN = tile >> 3;              // 0..15
  const int m16   = lane & 15;
  const int row   = tileM * 16 + m16;       // (b*T + p)
  const int col   = tileN * 16 + m16;       // d

  __builtin_prefetch(W_O + (size_t)lane * (DD * DD / 32), 0, 1);

  v8f acc = {};
  for (int kk = 0; kk < 8; ++kk) {
    v16h a16, b16;
    #pragma unroll
    for (int p = 0; p < 16; ++p) {
      a16[p] = (_Float16)zbuf[(size_t)row * (NN * HH) + kk * 32 + kmapA(lane, p)];
      b16[p] = (_Float16)W_O[(size_t)(kk * 32 + kmapB(lane, p)) * DD + col];
    }
    acc = __builtin_amdgcn_wmma_f32_16x16x32_f16(false, a16, false, b16,
                                                 (short)0, acc, false, false);
  }
  const float bias   = b_O[col];
  const int   rowOff = (lane >> 4) << 3;
  #pragma unroll
  for (int r = 0; r < 8; ++r) {
    int grow = tileM * 16 + rowOff + r;
    out[(size_t)grow * DD + col] = acc[r] + bias;
  }
}

extern "C" void kernel_launch(void* const* d_in, const int* in_sizes, int n_in,
                              void* d_out, int out_size, void* d_ws, size_t ws_size,
                              hipStream_t stream) {
  const float* x    = (const float*)d_in[0];
  const float* W_A  = (const float*)d_in[1];
  const float* W_Bp = (const float*)d_in[2];
  const float* W_C  = (const float*)d_in[3];
  const float* W_D  = (const float*)d_in[4];
  const float* W_DE = (const float*)d_in[5];
  const float* W_O  = (const float*)d_in[6];
  const float* b_A  = (const float*)d_in[7];
  const float* b_Bp = (const float*)d_in[8];
  const float* b_C  = (const float*)d_in[9];
  const float* b_D  = (const float*)d_in[10];
  const float* b_DE = (const float*)d_in[11];
  const float* b_O  = (const float*)d_in[12];
  float* out = (float*)d_out;

  float* proj = (float*)d_ws;                           // 7 * B*N*T*H f32 = 896 KB
  float* zbuf = proj + (size_t)7 * BB * NN * TT * HH;   // B*T*N*H f32   = 128 KB

  qa_proj_kernel<<<112, 256, 0, stream>>>(x, W_A, W_Bp, W_C, W_D, W_DE,
                                          b_A, b_Bp, b_C, b_D, proj);
  qa_attn_kernel<<<dim3(BB * NN, TT / 8), 256, 0, stream>>>(proj, b_DE, zbuf);
  qa_out_kernel<<<16, 256, 0, stream>>>(zbuf, W_O, b_O, out);
}
